// GCN_20289425506513
// MI455X (gfx1250) — matrix operand order repaired and checked by reading, not compiled, not run on hardware
//
#include <hip/hip_runtime.h>
#include <hip/hip_bf16.h>
#include <stdint.h>

// ---------- types for WMMA ----------
typedef __attribute__((ext_vector_type(16))) __bf16 v16bf;
typedef __attribute__((ext_vector_type(8)))  float  v8f;

__device__ __forceinline__ unsigned short f2bf_rne(float f) {
  unsigned u = __float_as_uint(f);
  u += 0x7fffu + ((u >> 16) & 1u);        // round-to-nearest-even
  return (unsigned short)(u >> 16);
}

__device__ __forceinline__ void atomAddF32(float* p, float v) {
  __hip_atomic_fetch_add(p, v, __ATOMIC_RELAXED, __HIP_MEMORY_SCOPE_AGENT);
}

// ---------- small prep kernels ----------
__global__ void k_cvt_bf16(const float* __restrict__ in,
                           unsigned short* __restrict__ out, int n) {
  int i = blockIdx.x * blockDim.x + threadIdx.x;
  if (i < n) out[i] = f2bf_rne(in[i]);
}

// W [K x Nor] f32  ->  Wt [Np x K] bf16 (transposed, zero-padded rows n>=Nor)
__global__ void k_cvt_wt(const float* __restrict__ W,
                         unsigned short* __restrict__ Wt,
                         int K, int Nor, int Np) {
  int i = blockIdx.x * blockDim.x + threadIdx.x;
  if (i >= Np * K) return;
  int n = i / K, k = i % K;
  float v = (n < Nor) ? W[k * Nor + n] : 0.0f;
  Wt[i] = f2bf_rne(v);
}

__global__ void k_fill1(float* __restrict__ a, float* __restrict__ b, int n) {
  int i = blockIdx.x * blockDim.x + threadIdx.x;
  if (i < n) { a[i] = 1.0f; b[i] = 1.0f; }   // self-loop contributes 1 to degree
}

__global__ void k_edge_deg(const int* __restrict__ col,
                           const float* __restrict__ ew,
                           float* __restrict__ degw, float* __restrict__ deg1, int E) {
  int e = blockIdx.x * blockDim.x + threadIdx.x;
  if (e >= E) return;
  int c = col[e];
  atomAddF32(&degw[c], ew[e]);
  atomAddF32(&deg1[c], 1.0f);
}

__global__ void k_rsqrt2(float* __restrict__ a, float* __restrict__ b, int n) {
  int i = blockIdx.x * blockDim.x + threadIdx.x;
  if (i >= n) return;
  float da = a[i], db = b[i];
  a[i] = da > 0.f ? rsqrtf(da) : 0.f;        // deg >= 1 always, but keep guard
  b[i] = db > 0.f ? rsqrtf(db) : 0.f;
}

__global__ void k_edge_norm(const int* __restrict__ row, const int* __restrict__ col,
                            const float* __restrict__ ew,
                            const float* __restrict__ disw, const float* __restrict__ dis1,
                            float* __restrict__ normw, float* __restrict__ norm1, int E) {
  int e = blockIdx.x * blockDim.x + threadIdx.x;
  if (e >= E) return;
  int r = row[e], c = col[e];
  normw[e] = disw[r] * ew[e] * disw[c];
  norm1[e] = dis1[r] * dis1[c];
}

// ---------- bf16 WMMA GEMM: C[M x ldc] = A[M x K] * Bt^T, Bt is [Npad x K] ----------
__global__ void __launch_bounds__(256) k_gemm_bf16(
    const unsigned short* __restrict__ A, const unsigned short* __restrict__ Bt,
    float* __restrict__ C, int M, int K, int ldc, int n_tiles, int n_real) {
  int wave = threadIdx.x >> 5;
  int lane = threadIdx.x & 31;
  int tile = blockIdx.x * 8 + wave;
  int tiles_total = (M >> 4) * n_tiles;
  if (tile >= tiles_total) return;               // uniform per wave
  int mtile = tile / n_tiles;
  int ntile = tile % n_tiles;
  int half  = lane >> 4;
  int lid   = lane & 15;
  int m0 = mtile * 16 + lid;                     // A row this lane holds
  int n0 = ntile * 16 + lid;                     // Bt row this lane holds
  const unsigned* Au = (const unsigned*)A;       // bf16 pairs as dwords
  const unsigned* Bu = (const unsigned*)Bt;
  size_t abase = ((size_t)m0 * K) >> 1;
  size_t bbase = ((size_t)n0 * K) >> 1;

  union FragA { v16bf v; unsigned u[8]; } fa;
  union FragB { v16bf v; unsigned u[8]; } fb;
  v8f c = {};

  for (int kk = 0; kk < K; kk += 32) {
#pragma unroll
    for (int v = 0; v < 8; ++v) {
      // A 16x32 bf16 layout: V0..3 -> K = half*8 + 2v ; V4..7 -> K = 16 + half*8 + 2(v-4)
      int ka = kk + ((v < 4) ? (half * 8 + v * 2) : (16 + half * 8 + (v - 4) * 2));
      // B 32x16 bf16 layout: V0..3 -> K = half*8 + 2v ; V4..7 -> K = 16 + half*8 + 2(v-4)
      int kb = kk + ((v < 4) ? (half * 8 + v * 2) : (16 + half * 8 + (v - 4) * 2));
      fa.u[v] = Au[abase + (ka >> 1)];
      fb.u[v] = Bu[bbase + (kb >> 1)];
    }
    c = __builtin_amdgcn_wmma_f32_16x16x32_bf16(
        false, fa.v, false, fb.v, (short)0, c, false, false);
  }

  if (n0 < n_real) {
#pragma unroll
    for (int r = 0; r < 8; ++r) {
      int mrow = mtile * 16 + r + half * 8;      // C/D layout: VGPR r -> M = r + 8*half
      C[(size_t)mrow * ldc + n0] = c[r];
    }
  }
}

// ---------- edge scatter: agg[col] += norm * h[row], F multiple of 4 ----------
__global__ void k_scatter(const float* __restrict__ h, const int* __restrict__ row,
                          const int* __restrict__ col, const float* __restrict__ norm,
                          float* __restrict__ agg, int E, int F) {
  int F4 = F >> 2;
  long i = (long)blockIdx.x * blockDim.x + threadIdx.x;
  if (i >= (long)E * F4) return;
  int e  = (int)(i / F4);
  int ch = (int)(i % F4);
  int r = row[e], c = col[e];
  float nw = norm[e];
  const float4 hv = *(const float4*)(h + (size_t)r * F + ch * 4);
  float* dst = agg + (size_t)c * F + ch * 4;
  atomAddF32(dst + 0, nw * hv.x);
  atomAddF32(dst + 1, nw * hv.y);
  atomAddF32(dst + 2, nw * hv.z);
  atomAddF32(dst + 3, nw * hv.w);
}

// ---------- finalize: out = [relu](agg + dis^2 * h + b); optional f32 / bf16 outputs ----------
__global__ void k_finalize(const float* __restrict__ agg, const float* __restrict__ h,
                           const float* __restrict__ dis, const float* __restrict__ bias,
                           float* __restrict__ out32, unsigned short* __restrict__ outbf,
                           int Nn, int F, int do_relu) {
  long i = (long)blockIdx.x * blockDim.x + threadIdx.x;
  if (i >= (long)Nn * F) return;
  int node = (int)(i / F);
  int f    = (int)(i % F);
  float d = dis[node];
  float v = agg[i] + d * d * h[i] + bias[f];
  if (do_relu) v = fmaxf(v, 0.0f);
  if (out32) out32[i] = v;
  if (outbf) outbf[i] = f2bf_rne(v);
}

// ============================================================================
extern "C" void kernel_launch(void* const* d_in, const int* in_sizes, int n_in,
                              void* d_out, int out_size, void* d_ws, size_t ws_size,
                              hipStream_t stream) {
  const int F_IN = 512, H = 128, C = 40, CP = 48;
  const float* x  = (const float*)d_in[0];
  const int*   ei = (const int*)d_in[1];
  const float* ew = (const float*)d_in[2];
  const float* W1 = (const float*)d_in[3];
  const float* b1 = (const float*)d_in[4];
  const float* W2 = (const float*)d_in[5];
  const float* b2 = (const float*)d_in[6];
  const float* W3 = (const float*)d_in[7];
  const float* b3 = (const float*)d_in[8];
  float* out = (float*)d_out;

  const int Nn = in_sizes[0] / F_IN;         // 100000 (multiple of 16)
  const int E  = in_sizes[1] / 2;            // 1.6M
  const int* row = ei;
  const int* col = ei + E;

  // ---- carve workspace ----
  char* p = (char*)d_ws;
  auto carve = [&](size_t bytes) { char* q = p; p += (bytes + 255) & ~size_t(255); return q; };
  unsigned short* xb   = (unsigned short*)carve((size_t)Nn * F_IN * 2);
  unsigned short* w1t  = (unsigned short*)carve((size_t)H  * F_IN * 2);
  unsigned short* w2t  = (unsigned short*)carve((size_t)H  * H * 2);
  unsigned short* w3t  = (unsigned short*)carve((size_t)CP * H * 2);
  float* disw  = (float*)carve((size_t)Nn * 4);
  float* dis1  = (float*)carve((size_t)Nn * 4);
  float* normw = (float*)carve((size_t)E * 4);
  float* norm1 = (float*)carve((size_t)E * 4);
  float* t     = (float*)carve((size_t)Nn * H * 4);   // GEMM output (pre-aggregation h)
  float* agg   = (float*)carve((size_t)Nn * H * 4);
  unsigned short* hb = (unsigned short*)carve((size_t)Nn * H * 2); // relu(h) in bf16

  const int T = 256;
  auto blocks = [](long n, int t) { return (unsigned)((n + t - 1) / t); };

  // ---- precision conversion ----
  k_cvt_bf16<<<blocks((long)Nn * F_IN, T), T, 0, stream>>>(x, xb, Nn * F_IN);
  k_cvt_wt<<<blocks((long)H * F_IN, T), T, 0, stream>>>(W1, w1t, F_IN, H, H);
  k_cvt_wt<<<blocks((long)H * H, T), T, 0, stream>>>(W2, w2t, H, H, H);
  k_cvt_wt<<<blocks((long)CP * H, T), T, 0, stream>>>(W3, w3t, H, C, CP);

  // ---- gcn_norm: degrees (init 1 for self loop) -> D^-1/2 -> per-edge coefficients ----
  k_fill1<<<blocks(Nn, T), T, 0, stream>>>(disw, dis1, Nn);
  k_edge_deg<<<blocks(E, T), T, 0, stream>>>(col, ew, disw, dis1, E);
  k_rsqrt2<<<blocks(Nn, T), T, 0, stream>>>(disw, dis1, Nn);
  k_edge_norm<<<blocks(E, T), T, 0, stream>>>(row, col, ew, disw, dis1, normw, norm1, E);

  const int mt = Nn >> 4;  // 6250 m-tiles

  // ---- layer 1: t = x @ W1 ; agg ; h1 = relu(agg + self + b1) -> bf16 ----
  k_gemm_bf16<<<blocks((long)mt * (H / 16), 8), T, 0, stream>>>(xb, w1t, t, Nn, F_IN, H, H / 16, H);
  hipMemsetAsync(agg, 0, (size_t)Nn * H * 4, stream);
  k_scatter<<<blocks((long)E * (H / 4), T), T, 0, stream>>>(t, row, col, normw, agg, E, H);
  k_finalize<<<blocks((long)Nn * H, T), T, 0, stream>>>(agg, t, disw, b1, nullptr, hb, Nn, H, 1);

  // ---- layer 2: t = h1 @ W2 ; agg ; h2 = relu(...) -> bf16 ----
  k_gemm_bf16<<<blocks((long)mt * (H / 16), 8), T, 0, stream>>>(hb, w2t, t, Nn, H, H, H / 16, H);
  hipMemsetAsync(agg, 0, (size_t)Nn * H * 4, stream);
  k_scatter<<<blocks((long)E * (H / 4), T), T, 0, stream>>>(t, row, col, normw, agg, E, H);
  k_finalize<<<blocks((long)Nn * H, T), T, 0, stream>>>(agg, t, disw, b2, nullptr, hb, Nn, H, 1);

  // ---- layer 3 (unit edge weights, no relu): out = agg + self + b3 ----
  k_gemm_bf16<<<blocks((long)mt * (CP / 16), 8), T, 0, stream>>>(hb, w3t, t, Nn, H, C, CP / 16, C);
  hipMemsetAsync(agg, 0, (size_t)Nn * C * 4, stream);
  k_scatter<<<blocks((long)E * (C / 4), T), T, 0, stream>>>(t, row, col, norm1, agg, E, C);
  k_finalize<<<blocks((long)Nn * C, T), T, 0, stream>>>(agg, t, dis1, b3, out, nullptr, Nn, C, 0);
}
